// VQuantizer_38062000177294
// MI455X (gfx1250) — matrix operand order repaired and checked by reading, not compiled, not run on hardware
//
#include <hip/hip_runtime.h>
#include <hip/hip_bf16.h>

typedef __attribute__((ext_vector_type(16))) _Float16     v16h;
typedef __attribute__((ext_vector_type(8)))  _Float16     v8h;
typedef __attribute__((ext_vector_type(8)))  float        v8f;
typedef __attribute__((ext_vector_type(4)))  unsigned int u32x4;
typedef __attribute__((ext_vector_type(8)))  int          i32x8;
typedef __attribute__((ext_vector_type(4)))  int          i32x4;

#define B_   16
#define D_   256
#define L_   2048
#define K_   8192
#define N_   (B_ * L_)      // 32768
#define KT_  (K_ / 16)      // 512 k-tiles
#define AROW 264            // padded row stride (halves): 528B rows, conflict-free frags

#define A_BYTES  (128 * AROW * 2)   // 67584
#define B_BYTES  (16 * AROW * 2)    // 8448 (16 rows x 512B + 16 x 16B TDM pad)
#define E2_OFF   (A_BYTES + 2 * B_BYTES)
#define SMEM_SZ  (E2_OFF + K_ * 4)

// ---------------- Kernel 1: codebook f32 -> f16, plus e2[k] = ||e_k||^2 ----------------
__global__ __launch_bounds__(256) void vq_prep(const float* __restrict__ cb,
                                               _Float16* __restrict__ cb16,
                                               float* __restrict__ e2) {
    __shared__ float red[256];
    const int k = blockIdx.x;
    const int d = threadIdx.x;
    float v = cb[(size_t)k * D_ + d];
    cb16[(size_t)k * D_ + d] = (_Float16)v;
    red[d] = v * v;
    __syncthreads();
    for (int s = 128; s > 0; s >>= 1) {
        if (d < s) red[d] += red[d + s];
        __syncthreads();
    }
    if (d == 0) e2[k] = red[0];
}

// Issue one TDM DMA: 8KB codebook tile (16 rows x 512B) -> LDS, with HW padding of
// 16B after every 512B row (pad_interval=6 ->128 DW, pad_amount=3 -> 4 DW), which
// reproduces the 528B padded row stride the WMMA fragment reads expect.
__device__ __forceinline__ void tdm_load_tile(unsigned lds_off, unsigned long long gaddr) {
    u32x4 g0;
    g0[0] = 1u;                                   // count=1, user descriptor
    g0[1] = lds_off;                              // LDS byte address
    g0[2] = (unsigned)(gaddr & 0xFFFFFFFFull);    // global_addr[31:0]
    g0[3] = (unsigned)((gaddr >> 32) & 0x01FFFFFFull) | (2u << 30);  // addr[56:32] | type=2

    i32x8 g1;
    g1[0] = (int)((1u << 16) |                    // data_size = 1 (2 bytes)
                  (1u << 20) |                    // pad_enable
                  (6u << 22) |                    // pad_interval: 128 DWORDs (512B)
                  (3u << 25));                    // pad_amount:   4 DWORDs (16B)
    g1[1] = (int)((4096u & 0xFFFFu) << 16);       // tensor_dim0[15:0] = 4096 (elements)
    g1[2] = (int)(1u << 16);                      // tensor_dim0[31:16]=0 | tensor_dim1=1
    g1[3] = (int)(4096u << 16);                   // tile_dim0 = 4096 elements (8KB)
    g1[4] = 1;                                    // tile_dim1 = 1, tile_dim2 = 0
    g1[5] = 4096;                                 // tensor_dim0_stride = 4096
    g1[6] = 0;
    g1[7] = 0;

    i32x4 gz4 = {0, 0, 0, 0};
    i32x8 gz8 = {0, 0, 0, 0, 0, 0, 0, 0};
    // clang-23 / therock-10.0 toolchain: 6-arg form
    __builtin_amdgcn_tensor_load_to_lds(g0, g1, gz4, gz4, gz8, 0);
}

// ---------------- Kernel 2: fused GEMM (z . cb^T) + argmin over K via WMMA + TDM ------
// Grid: N/128 blocks of 256 threads (8 waves). Wave w owns 16 n-rows; block sweeps all K.
// B tiles are DMA'd by the Tensor Data Mover into double-buffered LDS (1 barrier/iter).
__global__ __launch_bounds__(256) void vq_argmin(const float* __restrict__ z,
                                                 const _Float16* __restrict__ cb16,
                                                 const float* __restrict__ e2g,
                                                 int* __restrict__ idx) {
    extern __shared__ char smem[];
    _Float16* A   = (_Float16*)smem;              // 128 x AROW halves (z tile, f16)
    _Float16* Bs  = (_Float16*)(smem + A_BYTES);  // 2 x (16 x AROW) halves, ping-pong
    float*    e2s = (float*)(smem + E2_OFF);      // 8192 floats

    const int tid  = threadIdx.x;
    const int wave = tid >> 5;
    const int lane = tid & 31;
    const int hh   = lane >> 4;   // half-wave index (fragment butterfly)
    const int l15  = lane & 15;

    const int n0 = blockIdx.x * 128;     // 128 rows per block, never crosses a batch
    const int b  = n0 / L_;
    const int l0 = n0 % L_;

    const unsigned long long cbg = (unsigned long long)(const void*)cb16;

    // Kick off DMA of B tile 0 into buffer 0 while we stage A (disjoint LDS regions).
    if (wave == 0) tdm_load_tile(A_BYTES, cbg);

    // Stage A tile: A[l][d] = (f16) z[b][d][l0+l]  (coalesced along l)
    {
        const float* zb = z + (size_t)b * D_ * L_ + l0;
        for (int i = 0; i < 128; ++i) {
            int f  = tid + i * 256;      // 0..32767
            int d  = f >> 7;
            int lo = f & 127;
            A[lo * AROW + d] = (_Float16)zb[(size_t)d * L_ + lo];
        }
        for (int i = tid; i < K_; i += 256) e2s[i] = e2g[i];
    }
    __syncthreads();

    // Resident A fragments: 8 x v16h per lane (16 rows x 256 d per wave)
    // ISA layout: lane<16 -> K {c*32+0..7, c*32+16..23}; lane>=16 -> {+8..15, +24..31}
    v16h afrag[8];
    {
        const _Float16* ar = A + (wave * 16 + l15) * AROW;
#pragma unroll
        for (int c = 0; c < 8; ++c) {
            int d0 = c * 32 + hh * 8;
            v8h lo = *(const v8h*)(ar + d0);
            v8h hi = *(const v8h*)(ar + d0 + 16);
            v16h a;
            *((v8h*)&a)     = lo;
            *((v8h*)&a + 1) = hi;
            afrag[c] = a;
        }
    }

    float bestv[8];
    int   besti[8];
#pragma unroll
    for (int j = 0; j < 8; ++j) { bestv[j] = 3.4e38f; besti[j] = 0; }

    for (int t = 0; t < KT_; ++t) {
        // Tile t's DMA (issued last iteration / prologue) must have landed.
        if (wave == 0) __builtin_amdgcn_s_wait_tensorcnt(0);
        // One barrier: publishes tile t to all waves AND guarantees everyone is done
        // reading the buffer tile t+1 will overwrite (used at iteration t-1).
        __syncthreads();
        if (wave == 0 && (t + 1) < KT_) {
            unsigned buf = ((unsigned)(t + 1) & 1u) * B_BYTES;
            tdm_load_tile(A_BYTES + buf, cbg + (unsigned long long)(t + 1) * 16 * D_ * 2);
        }

        // 16x16 ez tile, accumulate over D=256 (8 chained WMMAs)
        v8f acc = {};
        const _Float16* br = Bs + (unsigned)(t & 1) * (16 * AROW) + l15 * AROW;
#pragma unroll
        for (int c = 0; c < 8; ++c) {
            int d0 = c * 32 + hh * 8;           // B halves: {0..7,16..23} / {8..15,24..31}
            v8h lo = *(const v8h*)(br + d0);
            v8h hi = *(const v8h*)(br + d0 + 16);
            v16h bfrag;
            *((v8h*)&bfrag)     = lo;
            *((v8h*)&bfrag + 1) = hi;
            acc = __builtin_amdgcn_wmma_f32_16x16x32_f16(
                false, afrag[c], false, bfrag, (short)0, acc, false, false);
        }

        // dist(k) relative part = e2[k] - 2*ez ; lane owns one k per tile, 8 rows
        int   kk = t * 16 + l15;
        float ev = e2s[kk];
#pragma unroll
        for (int j = 0; j < 8; ++j) {
            float dj = ev - 2.0f * acc[j];
            if (dj < bestv[j]) { bestv[j] = dj; besti[j] = kk; }
        }
    }

    // argmin across the 16 lanes of each half-wave (tie -> lowest index)
#pragma unroll
    for (int j = 0; j < 8; ++j) {
        float v = bestv[j]; int i = besti[j];
#pragma unroll
        for (int m = 1; m < 16; m <<= 1) {
            float v2 = __shfl_xor(v, m, 32);
            int   i2 = __shfl_xor(i, m, 32);
            if (v2 < v || (v2 == v && i2 < i)) { v = v2; i = i2; }
        }
        bestv[j] = v; besti[j] = i;
    }
    if (l15 == 0) {     // lane 0 -> rows 0..7, lane 16 -> rows 8..15
        int nb = n0 + wave * 16 + hh * 8;
#pragma unroll
        for (int j = 0; j < 8; ++j) idx[nb + j] = besti[j];
    }
}

// ---------------- Kernel 3: gather z_q (transposed) + per-block squared-error partials ----
// Grid: B*D*L/256 blocks; block = one (b,d) row chunk of 256 l's (coalesced out writes)
__global__ __launch_bounds__(256) void vq_gather(const float* __restrict__ z,
                                                 const float* __restrict__ cb,
                                                 const int* __restrict__ idx,
                                                 float* __restrict__ out,
                                                 float* __restrict__ partial) {
    __shared__ float red[256];
    const int blk    = blockIdx.x;
    const int lchunk = blk & 7;          // L/256 = 8 chunks
    const int bd     = blk >> 3;         // b*D + d
    const int b      = bd >> 8;
    const int d      = bd & 255;
    const int l      = lchunk * 256 + threadIdx.x;

    const size_t off = (size_t)bd * L_ + l;
    const int    n   = b * L_ + l;
    const int    k   = idx[n];
    const float  q   = cb[(size_t)k * D_ + d];
    const float  zv  = z[off];
    out[off] = q;                        // straight-through output == z_q value

    float diff = q - zv;
    red[threadIdx.x] = diff * diff;
    __syncthreads();
    for (int s = 128; s > 0; s >>= 1) {
        if (threadIdx.x < s) red[threadIdx.x] += red[threadIdx.x + s];
        __syncthreads();
    }
    if (threadIdx.x == 0) partial[blk] = red[0];
}

// ---------------- Kernel 4: deterministic loss reduction ----------------
__global__ __launch_bounds__(256) void vq_loss(const float* __restrict__ partial,
                                               float* __restrict__ out_loss) {
    __shared__ float red[256];
    float s = 0.f;
    for (int i = threadIdx.x; i < (B_ * D_ * L_) / 256; i += 256) s += partial[i];
    red[threadIdx.x] = s;
    __syncthreads();
    for (int st = 128; st > 0; st >>= 1) {
        if (threadIdx.x < st) red[threadIdx.x] += red[threadIdx.x + st];
        __syncthreads();
    }
    // commitment_loss == embedding_loss numerically -> vq_loss = 1.25 * MSE
    if (threadIdx.x == 0)
        out_loss[0] = red[0] * (1.25f / ((float)N_ * (float)D_));
}

extern "C" void kernel_launch(void* const* d_in, const int* in_sizes, int n_in,
                              void* d_out, int out_size, void* d_ws, size_t ws_size,
                              hipStream_t stream) {
    const float* z  = (const float*)d_in[0];   // [B, D, L] f32
    const float* cb = (const float*)d_in[1];   // [K, D]    f32
    float*       out = (float*)d_out;          // [B, D, L] f32 ++ [1] loss

    char* ws = (char*)d_ws;
    _Float16* cb16    = (_Float16*)ws;                                 // 4 MB
    float*    e2      = (float*)(ws + (size_t)K_ * D_ * sizeof(_Float16));
    int*      idx     = (int*)((char*)e2 + (size_t)K_ * sizeof(float));
    float*    partial = (float*)((char*)idx + (size_t)N_ * sizeof(int));

    vq_prep<<<K_, 256, 0, stream>>>(cb, cb16, e2);

    vq_argmin<<<N_ / 128, 256, SMEM_SZ, stream>>>(z, cb16, e2, idx);

    vq_gather<<<(B_ * D_ * L_) / 256, 256, 0, stream>>>(z, cb, idx, out, partial);

    vq_loss<<<1, 256, 0, stream>>>(partial, out + (size_t)B_ * D_ * L_);
}
